// Encoderlayer_71803263254630
// MI455X (gfx1250) — matrix operand order repaired and checked by reading, not compiled, not run on hardware
//
#include <hip/hip_runtime.h>

// ---------------------------------------------------------------------------
// Transformer encoder layer for MI455X (gfx1250, wave32, WMMA bf16).
// S=4096, D=1024, H=16, dk=64, F=4096.  Compute-bound (~172 GFLOP) -> bf16
// WMMA with fp32 accumulation; tiles staged via double-buffered
// global_load_async_to_lds_b128 (ASYNCcnt) so DMA overlaps WMMA compute.
// ---------------------------------------------------------------------------

#define S_LEN   4096
#define D_MODEL 1024
#define N_HEADS 16
#define D_HEAD  64
#define D_FF    4096

typedef __attribute__((ext_vector_type(16))) __bf16 bf16x16;
typedef __attribute__((ext_vector_type(8)))  float  f32x8;

union BF16Frag {
    bf16x16      v;
    unsigned int w[8];
};

// float -> bf16 (round to nearest even), raw ushort storage
__device__ __forceinline__ unsigned short f2bf(float f) {
    unsigned int u = __float_as_uint(f);
    u += 0x7FFFu + ((u >> 16) & 1u);
    return (unsigned short)(u >> 16);
}

// A-matrix (16x32 bf16) K index pattern: VGPR pair v (elements 2v,2v+1),
// lane half hi.  v<4: K=8*hi+2v ; v>=4: K=16+8*hi+2*(v-4)
__device__ __forceinline__ int a_koff(int v, int hi) {
    return (v < 4) ? (8 * hi + 2 * v) : (16 + 8 * hi + 2 * (v - 4));
}
// B-matrix (32x16 bf16) K index pattern (same as A)
__device__ __forceinline__ int b_koff(int v, int hi) {
    return (v < 4) ? (8 * hi + 2 * v) : (16 + 8 * hi + 2 * (v - 4));
}

// gfx1250 async global->LDS copy (ASYNCcnt-tracked), GVS addressing:
// mem = SADDR(64) + VADDR(32) ; LDS dest byte offset in VDST VGPR.
__device__ __forceinline__ void async_load_b128(unsigned lds_off,
                                                unsigned gl_off,
                                                const void* sbase) {
    asm volatile("global_load_async_to_lds_b128 %0, %1, %2"
                 :
                 : "v"(lds_off), "v"(gl_off), "s"(sbase)
                 : "memory");
}
__device__ __forceinline__ void wait_async0() {
    asm volatile("s_wait_asynccnt 0" ::: "memory");
}

// ---------------------------------------------------------------------------
// f32 -> bf16 conversion
// ---------------------------------------------------------------------------
__global__ void cvt_f32_to_bf16(const float* __restrict__ src,
                                unsigned short* __restrict__ dst, int n) {
    int i = blockIdx.x * blockDim.x + threadIdx.x;
    if (i < n) dst[i] = f2bf(src[i]);
}

// ---------------------------------------------------------------------------
// Custom LayerNorm: alpha*(x-mu)/(sqrt(var_unbiased)+eps)+beta, out bf16.
// One block (256 threads) per row of 1024.
// ---------------------------------------------------------------------------
__global__ __launch_bounds__(256) void layernorm_to_bf16(
    const float* __restrict__ x, unsigned short* __restrict__ out,
    const float* __restrict__ alpha_p, const float* __restrict__ beta_p) {
    __shared__ float red[256];
    const int row = blockIdx.x;
    const int tid = threadIdx.x;
    const float* xr = x + row * D_MODEL;

    float v[4];
    float s = 0.f;
#pragma unroll
    for (int i = 0; i < 4; i++) { v[i] = xr[tid + 256 * i]; s += v[i]; }
    red[tid] = s;
    __syncthreads();
    for (int off = 128; off > 0; off >>= 1) {
        if (tid < off) red[tid] += red[tid + off];
        __syncthreads();
    }
    const float mu = red[0] * (1.0f / (float)D_MODEL);
    __syncthreads();

    float sq = 0.f;
#pragma unroll
    for (int i = 0; i < 4; i++) { float d = v[i] - mu; sq += d * d; }
    red[tid] = sq;
    __syncthreads();
    for (int off = 128; off > 0; off >>= 1) {
        if (tid < off) red[tid] += red[tid + off];
        __syncthreads();
    }
    const float var = red[0] * (1.0f / (float)(D_MODEL - 1));
    const float inv = 1.0f / (sqrtf(var) + 1e-6f);
    const float a = alpha_p[0], b = beta_p[0];

    unsigned short* orow = out + row * D_MODEL;
#pragma unroll
    for (int i = 0; i < 4; i++)
        orow[tid + 256 * i] = f2bf(a * (v[i] - mu) * inv + b);
}

// ---------------------------------------------------------------------------
// bf16 GEMM: C[M,N] = A[M,K] * B[K,N] + bias (+residual / relu), WMMA bf16.
// Block: 256 threads (8 waves), tile 128x128, BK=32. Each wave: 32x64
// (2x4 WMMA accum tiles; A fragments reused across 4 B fragments).
// Double-buffered async staging: DMA of tile k+1 overlaps WMMA of tile k.
// ---------------------------------------------------------------------------
#define BM 128
#define BN 128
#define BK 32
#define A_PAD 8   // row stride 40 ushorts = 80 B (16B-aligned dests)
#define B_PAD 8   // row stride 136 ushorts = 272 B

#define EPI_BF16      0
#define EPI_BF16_RELU 1
#define EPI_F32_RES   2

template <int EPI>
__global__ __launch_bounds__(256) void gemm_bf16_kernel(
    const unsigned short* __restrict__ A, const unsigned short* __restrict__ B,
    const float* __restrict__ bias, const float* __restrict__ residual,
    float* __restrict__ outF, unsigned short* __restrict__ outB,
    int M, int N, int K) {
    __shared__ __align__(16) unsigned short As[2][BM][BK + A_PAD];
    __shared__ __align__(16) unsigned short Bs[2][BK][BN + B_PAD];

    const int tid  = threadIdx.x;
    const int lane = tid & 31;
    const int w    = tid >> 5;      // wave 0..7
    const int wrow = w >> 1;        // 0..3 -> 32 rows each
    const int wcol = w & 1;         // 0..1 -> 64 cols each
    const int hi   = lane >> 4;
    const int ln   = lane & 15;

    const int m0 = blockIdx.y * BM;
    const int n0 = blockIdx.x * BN;

    f32x8 acc[2][4];
#pragma unroll
    for (int i = 0; i < 2; i++)
#pragma unroll
        for (int j = 0; j < 4; j++) acc[i][j] = (f32x8)0.0f;

    // cooperative async loaders (per-thread 32B for A, 32B for B)
    const int arow = tid >> 1;          // 0..127
    const int acol = (tid & 1) * 16;    // 0 or 16
    const int brow = tid >> 3;          // 0..31
    const int bcol = (tid & 7) * 16;    // 0..112

    const unsigned a_lds0 = (unsigned)(size_t)&As[0][arow][acol];
    const unsigned b_lds0 = (unsigned)(size_t)&Bs[0][brow][bcol];
    const unsigned a_pl = (unsigned)sizeof(As[0]);
    const unsigned b_pl = (unsigned)sizeof(Bs[0]);

    auto issue_tile = [&](int buf, int k0) {
        unsigned a_go = 2u * (unsigned)((m0 + arow) * K + k0 + acol);
        unsigned b_go = 2u * (unsigned)((k0 + brow) * N + n0 + bcol);
        unsigned al = a_lds0 + (unsigned)buf * a_pl;
        unsigned bl = b_lds0 + (unsigned)buf * b_pl;
        async_load_b128(al,      a_go,      A);
        async_load_b128(al + 16, a_go + 16, A);
        async_load_b128(bl,      b_go,      B);
        async_load_b128(bl + 16, b_go + 16, B);
    };

    auto compute_tile = [&](int buf) {
        BF16Frag fa[2];
#pragma unroll
        for (int mi = 0; mi < 2; mi++) {
            const int m = wrow * 32 + mi * 16 + ln;
#pragma unroll
            for (int v = 0; v < 8; v++)
                fa[mi].w[v] = *(const unsigned int*)&As[buf][m][a_koff(v, hi)];
        }
#pragma unroll
        for (int ni = 0; ni < 4; ni++) {
            BF16Frag fb;
            const int n = wcol * 64 + ni * 16 + ln;
#pragma unroll
            for (int v = 0; v < 8; v++) {
                const int ko = b_koff(v, hi);
                unsigned int lo = Bs[buf][ko][n];
                unsigned int h2 = Bs[buf][ko + 1][n];
                fb.w[v] = lo | (h2 << 16);
            }
#pragma unroll
            for (int mi = 0; mi < 2; mi++)
                acc[mi][ni] = __builtin_amdgcn_wmma_f32_16x16x32_bf16(
                    false, fa[mi].v, false, fb.v, (short)0, acc[mi][ni],
                    false, false);
        }
    };

    // prologue: stage tile 0 into buffer 0
    issue_tile(0, 0);
    wait_async0();
    __syncthreads();

    // K is a multiple of 64 for every call site
    for (int k0 = 0; k0 < K; k0 += 2 * BK) {
        if (k0 + BK < K) issue_tile(1, k0 + BK);     // DMA next, overlap
        compute_tile(0);
        wait_async0();
        __syncthreads();
        if (k0 + 2 * BK < K) issue_tile(0, k0 + 2 * BK);
        compute_tile(1);
        wait_async0();
        __syncthreads();
    }

    // epilogue: C/D layout -> lane holds col n=lane&15, rows m=r+8*hi
#pragma unroll
    for (int mi = 0; mi < 2; mi++) {
#pragma unroll
        for (int ni = 0; ni < 4; ni++) {
            const int col = n0 + wcol * 64 + ni * 16 + ln;
            const float bv = bias[col];
#pragma unroll
            for (int r = 0; r < 8; r++) {
                const int row = m0 + wrow * 32 + mi * 16 + r + 8 * hi;
                float val = acc[mi][ni][r] + bv;
                if (EPI == EPI_BF16_RELU) val = fmaxf(val, 0.0f);
                if (EPI == EPI_F32_RES) {
                    outF[row * N + col] = val + residual[row * N + col];
                } else {
                    outB[row * N + col] = f2bf(val);
                }
            }
        }
    }
}

// ---------------------------------------------------------------------------
// Flash-style attention. Grid (S/64, H), block 128 (4 waves).
// Each wave: 16 query rows. Block stages K/V 32-key tiles in LDS with
// double-buffered async DMA; online softmax; ctx accumulated with WMMA.
// ---------------------------------------------------------------------------
#define KV_PAD 8   // row stride 72 ushorts = 144 B (16B-aligned dests)

__global__ __launch_bounds__(128) void attention_kernel(
    const unsigned short* __restrict__ Qb, const unsigned short* __restrict__ Kb,
    const unsigned short* __restrict__ Vb, unsigned short* __restrict__ Ob) {
    __shared__ __align__(16) unsigned short Ks[2][32][D_HEAD + KV_PAD];
    __shared__ __align__(16) unsigned short Vs[2][32][D_HEAD + KV_PAD];
    __shared__ unsigned short Ps[4][16][36];   // per-wave P [16 q x 32 k]

    const int tid  = threadIdx.x;
    const int lane = tid & 31;
    const int w    = tid >> 5;                  // 0..3
    const int hi   = lane >> 4;
    const int ln   = lane & 15;
    const int h    = blockIdx.y;
    const int q0   = blockIdx.x * 64 + w * 16;
    const int hoff = h * D_HEAD;
    const float scale = 0.125f;                 // 1/sqrt(64)

    // staging indices: 128 threads cover 32x64 ushorts, 16 per thread
    const int skey = tid >> 2;                  // 0..31
    const int sd   = (tid & 3) * 16;            // 0..48

    const unsigned k_lds0 = (unsigned)(size_t)&Ks[0][skey][sd];
    const unsigned v_lds0 = (unsigned)(size_t)&Vs[0][skey][sd];
    const unsigned kv_pl  = (unsigned)sizeof(Ks[0]);

    auto issue_kv = [&](int buf, int j) {
        unsigned go = 2u * (unsigned)((j + skey) * D_MODEL + hoff + sd);
        unsigned kl = k_lds0 + (unsigned)buf * kv_pl;
        unsigned vl = v_lds0 + (unsigned)buf * kv_pl;
        async_load_b128(kl,      go,      Kb);
        async_load_b128(kl + 16, go + 16, Kb);
        async_load_b128(vl,      go,      Vb);
        async_load_b128(vl + 16, go + 16, Vb);
    };

    // Q fragments: two 16x32 A-tiles covering dk=64
    BF16Frag qf[2];
#pragma unroll
    for (int dh = 0; dh < 2; dh++) {
        const unsigned short* qrow = Qb + (q0 + ln) * D_MODEL + hoff + dh * 32;
#pragma unroll
        for (int v = 0; v < 8; v++)
            qf[dh].w[v] = *(const unsigned int*)&qrow[a_koff(v, hi)];
    }

    f32x8 acc_o[4];
#pragma unroll
    for (int c = 0; c < 4; c++) acc_o[c] = (f32x8)0.0f;
    float rmax[8], rsum[8];
#pragma unroll
    for (int r = 0; r < 8; r++) { rmax[r] = -1e30f; rsum[r] = 0.f; }

    auto step = [&](int buf) {
        // scores for 32 keys: two 16-key subtiles, K-dim = dk = 64
        f32x8 sc[2];
#pragma unroll
        for (int kt = 0; kt < 2; kt++) {
            sc[kt] = (f32x8)0.0f;
#pragma unroll
            for (int dh = 0; dh < 2; dh++) {
                BF16Frag kf;
#pragma unroll
                for (int v = 0; v < 8; v++)
                    kf.w[v] = *(const unsigned int*)
                        &Ks[buf][kt * 16 + ln][dh * 32 + b_koff(v, hi)];
                sc[kt] = __builtin_amdgcn_wmma_f32_16x16x32_bf16(
                    false, qf[dh].v, false, kf.v, (short)0, sc[kt], false, false);
            }
        }

        // online softmax (rows m = r+8*hi, key-cols across 16-lane halves)
#pragma unroll
        for (int r = 0; r < 8; r++) {
            float s0 = sc[0][r] * scale;
            float s1 = sc[1][r] * scale;
            float lm = fmaxf(s0, s1);
#pragma unroll
            for (int off = 1; off < 16; off <<= 1)
                lm = fmaxf(lm, __shfl_xor(lm, off));
            const float mnew = fmaxf(rmax[r], lm);
            const float corr = __expf(rmax[r] - mnew);
            rmax[r] = mnew;
            const float p0 = __expf(s0 - mnew);
            const float p1 = __expf(s1 - mnew);
            float ls = p0 + p1;
#pragma unroll
            for (int off = 1; off < 16; off <<= 1)
                ls += __shfl_xor(ls, off);
            rsum[r] = rsum[r] * corr + ls;
#pragma unroll
            for (int c = 0; c < 4; c++) acc_o[c][r] *= corr;
            // stage P (C-layout -> LDS row-major [q][key])
            Ps[w][r + 8 * hi][ln]      = f2bf(p0);
            Ps[w][r + 8 * hi][16 + ln] = f2bf(p1);
        }
        __syncthreads();

        // ctx += P[16x32] * V[32 x 64]  (4 WMMAs over d-chunks)
        BF16Frag pf;
#pragma unroll
        for (int v = 0; v < 8; v++)
            pf.w[v] = *(const unsigned int*)&Ps[w][ln][a_koff(v, hi)];
#pragma unroll
        for (int c = 0; c < 4; c++) {
            BF16Frag vf;
            const int dcol = c * 16 + ln;
#pragma unroll
            for (int v = 0; v < 8; v++) {
                const int kd = b_koff(v, hi);
                unsigned int lo = Vs[buf][kd][dcol];
                unsigned int h2 = Vs[buf][kd + 1][dcol];
                vf.w[v] = lo | (h2 << 16);
            }
            acc_o[c] = __builtin_amdgcn_wmma_f32_16x16x32_bf16(
                false, pf.v, false, vf.v, (short)0, acc_o[c], false, false);
        }
    };

    // prologue: stage key tile 0 into buffer 0
    issue_kv(0, 0);
    wait_async0();
    __syncthreads();

    for (int j = 0; j < S_LEN; j += 64) {
        if (j + 32 < S_LEN) issue_kv(1, j + 32);     // DMA next, overlap
        step(0);
        wait_async0();
        __syncthreads();
        if (j + 64 < S_LEN) issue_kv(0, j + 64);
        step(1);
        wait_async0();
        __syncthreads();
    }

    // normalize and write ctx (bf16, [S, D] with head offset)
#pragma unroll
    for (int r = 0; r < 8; r++) {
        const float inv = 1.0f / rsum[r];
        const int row = q0 + r + 8 * hi;
#pragma unroll
        for (int c = 0; c < 4; c++)
            Ob[row * D_MODEL + hoff + c * 16 + ln] = f2bf(acc_o[c][r] * inv);
    }
}

// ---------------------------------------------------------------------------
// Host-side launch
// ---------------------------------------------------------------------------
extern "C" void kernel_launch(void* const* d_in, const int* in_sizes, int n_in,
                              void* d_out, int out_size, void* d_ws, size_t ws_size,
                              hipStream_t stream) {
    const float* x    = (const float*)d_in[0];
    const float* Wq   = (const float*)d_in[1];
    const float* bq   = (const float*)d_in[2];
    const float* Wk   = (const float*)d_in[3];
    const float* bk   = (const float*)d_in[4];
    const float* Wv   = (const float*)d_in[5];
    const float* bv   = (const float*)d_in[6];
    const float* Wo   = (const float*)d_in[7];
    const float* bo   = (const float*)d_in[8];
    const float* ln1a = (const float*)d_in[9];
    const float* ln1b = (const float*)d_in[10];
    const float* W1   = (const float*)d_in[11];
    const float* b1   = (const float*)d_in[12];
    const float* W2   = (const float*)d_in[13];
    const float* b2   = (const float*)d_in[14];
    const float* ln2a = (const float*)d_in[15];
    const float* ln2b = (const float*)d_in[16];

    const int S = S_LEN, Dm = D_MODEL, F = D_FF;

    char* ws = (char*)d_ws;
    size_t off = 0;
    auto alloc = [&](size_t bytes) -> void* {
        void* p = ws + off;
        off += (bytes + 255) & ~(size_t)255;
        return p;
    };
    unsigned short* h1_bf  = (unsigned short*)alloc((size_t)S * Dm * 2);
    unsigned short* wq_bf  = (unsigned short*)alloc((size_t)Dm * Dm * 2);
    unsigned short* wk_bf  = (unsigned short*)alloc((size_t)Dm * Dm * 2);
    unsigned short* wv_bf  = (unsigned short*)alloc((size_t)Dm * Dm * 2);
    unsigned short* wo_bf  = (unsigned short*)alloc((size_t)Dm * Dm * 2);
    unsigned short* w1_bf  = (unsigned short*)alloc((size_t)Dm * F * 2);
    unsigned short* w2_bf  = (unsigned short*)alloc((size_t)F * Dm * 2);
    unsigned short* q_bf   = (unsigned short*)alloc((size_t)S * Dm * 2);
    unsigned short* k_bf   = (unsigned short*)alloc((size_t)S * Dm * 2);
    unsigned short* v_bf   = (unsigned short*)alloc((size_t)S * Dm * 2);
    unsigned short* ctx_bf = (unsigned short*)alloc((size_t)S * Dm * 2);
    unsigned short* h2_bf  = (unsigned short*)alloc((size_t)S * Dm * 2);
    unsigned short* f1_bf  = (unsigned short*)alloc((size_t)S * F * 2);
    float*          xres1  = (float*)alloc((size_t)S * Dm * 4);
    (void)ws_size; (void)n_in; (void)in_sizes; (void)out_size;

    // weight conversions f32 -> bf16
    cvt_f32_to_bf16<<<(Dm * Dm + 255) / 256, 256, 0, stream>>>(Wq, wq_bf, Dm * Dm);
    cvt_f32_to_bf16<<<(Dm * Dm + 255) / 256, 256, 0, stream>>>(Wk, wk_bf, Dm * Dm);
    cvt_f32_to_bf16<<<(Dm * Dm + 255) / 256, 256, 0, stream>>>(Wv, wv_bf, Dm * Dm);
    cvt_f32_to_bf16<<<(Dm * Dm + 255) / 256, 256, 0, stream>>>(Wo, wo_bf, Dm * Dm);
    cvt_f32_to_bf16<<<(Dm * F + 255) / 256, 256, 0, stream>>>(W1, w1_bf, Dm * F);
    cvt_f32_to_bf16<<<(F * Dm + 255) / 256, 256, 0, stream>>>(W2, w2_bf, F * Dm);

    // LN1
    layernorm_to_bf16<<<S, 256, 0, stream>>>(x, h1_bf, ln1a, ln1b);

    // Q/K/V projections
    dim3 gQKV(Dm / BN, S / BM);
    gemm_bf16_kernel<EPI_BF16><<<gQKV, 256, 0, stream>>>(
        h1_bf, wq_bf, bq, nullptr, nullptr, q_bf, S, Dm, Dm);
    gemm_bf16_kernel<EPI_BF16><<<gQKV, 256, 0, stream>>>(
        h1_bf, wk_bf, bk, nullptr, nullptr, k_bf, S, Dm, Dm);
    gemm_bf16_kernel<EPI_BF16><<<gQKV, 256, 0, stream>>>(
        h1_bf, wv_bf, bv, nullptr, nullptr, v_bf, S, Dm, Dm);

    // attention
    dim3 gAttn(S / 64, N_HEADS);
    attention_kernel<<<gAttn, 128, 0, stream>>>(q_bf, k_bf, v_bf, ctx_bf);

    // output projection + residual 1 (fp32)
    gemm_bf16_kernel<EPI_F32_RES><<<gQKV, 256, 0, stream>>>(
        ctx_bf, wo_bf, bo, x, xres1, nullptr, S, Dm, Dm);

    // LN2
    layernorm_to_bf16<<<S, 256, 0, stream>>>(xres1, h2_bf, ln2a, ln2b);

    // FFN1 (+ReLU)
    dim3 gF1(F / BN, S / BM);
    gemm_bf16_kernel<EPI_BF16_RELU><<<gF1, 256, 0, stream>>>(
        h2_bf, w1_bf, b1, nullptr, nullptr, f1_bf, S, F, Dm);

    // FFN2 + residual 2 -> final fp32 output
    gemm_bf16_kernel<EPI_F32_RES><<<gQKV, 256, 0, stream>>>(
        f1_bf, w2_bf, b2, xres1, (float*)d_out, nullptr, S, Dm, F);
}
